// TemporalAttention_69947837382752
// MI455X (gfx1250) — matrix operand order repaired and checked by reading, not compiled, not run on hardware
//
#include <hip/hip_runtime.h>

// Problem constants (match reference)
#define B_  8
#define T_  64
#define N_  512
#define D_  64
#define KH  8      // heads
#define DH  8      // per-head dim

typedef __attribute__((ext_vector_type(16))) _Float16 v16h;
typedef __attribute__((ext_vector_type(8)))  _Float16 v8h;
typedef __attribute__((ext_vector_type(4)))  _Float16 v4h;
typedef __attribute__((ext_vector_type(8)))  float    v8f;

// Padded pitches (halves) for transposed weight tiles: keeps the strided
// transpose writes spread across LDS banks while preserving 16B alignment
// for v8h fragment loads (136*2 and 72*2 are multiples of 16).
constexpr int WP = 136;   // pitch for 64 x 128 transposed weights
constexpr int FP = 72;    // pitch for 64 x 64  transposed weights

// ---------------- LDS layout (units: halves) ----------------
constexpr int SH_H   = 0;                  // h   : 64 x 128 row-major
constexpr int SH_WQT = SH_H   + 64*128;    // WqT : 64 x WP (N-major, padded)
constexpr int SH_WKT = SH_WQT + 64*WP;
constexpr int SH_WVT = SH_WKT + 64*WP;
constexpr int SH_W1T = SH_WVT + 64*WP;     // W1T : 64 x FP
constexpr int SH_W2T = SH_W1T + 64*FP;
constexpr int SH_Q   = SH_W2T + 64*FP;     // q   : 64 x 64 row-major
constexpr int SH_K   = SH_Q   + 64*64;     // k   : 64 x 64 row-major
constexpr int SH_VT  = SH_K   + 64*64;     // vT  : 64 x 64 (feature-major)
constexpr int SH_O   = SH_VT  + 64*64;     // attn out, row-major
constexpr int SH_O1  = SH_O   + 64*64;     // ffn hidden, row-major
constexpr int SH_P   = SH_O1  + 64*64;     // probs: 8 slots of 64x64 (per wave)
constexpr int SH_END = SH_P   + 8*64*64;
constexpr size_t SMEM_BYTES = (size_t)SH_END * 2 + 5 * 64 * 4;  // + biases (f32)

// ---------------- fragment helpers ----------------
// A fragment: 16x32 f16 from row-major LDS matrix (pitch in halves).
// Lane L: row = L%16; chunk0 K = 8*(L/16)+0..7, chunk1 K = 16+8*(L/16)+0..7.
__device__ __forceinline__ v16h load_A(const _Float16* __restrict__ s, int pitch) {
  const int lane = threadIdx.x & 31;
  const _Float16* p = s + (lane & 15) * pitch + 8 * (lane >> 4);
  v8h lo = *(const v8h*)(p);
  v8h hi = *(const v8h*)(p + 16);
  return __builtin_shufflevector(lo, hi, 0,1,2,3,4,5,6,7,8,9,10,11,12,13,14,15);
}

// B fragment: 32x16 f16 read from Bt stored N-major (row n = column of B),
// pitch in halves. Lane L: N = L%16; chunk0 K = 8*(L/16)+0..7, chunk1 K = 16+8*(L/16)+0..7.
__device__ __forceinline__ v16h load_Bt(const _Float16* __restrict__ bt, int pitch) {
  const int lane = threadIdx.x & 31;
  const _Float16* p = bt + (lane & 15) * pitch + 8 * (lane >> 4);
  v8h lo = *(const v8h*)(p);
  v8h hi = *(const v8h*)(p + 16);
  return __builtin_shufflevector(lo, hi, 0,1,2,3,4,5,6,7,8,9,10,11,12,13,14,15);
}

__device__ __forceinline__ v8f wmma16(v16h a, v16h b, v8f c) {
  return __builtin_amdgcn_wmma_f32_16x16x32_f16(false, a, false, b, (short)0, c,
                                                false, false);
}

__global__ void __launch_bounds__(256)
ta_kernel(const float* __restrict__ x,  const float* __restrict__ ste,
          const float* __restrict__ Wq, const float* __restrict__ bq,
          const float* __restrict__ Wk, const float* __restrict__ bk,
          const float* __restrict__ Wv, const float* __restrict__ bv,
          const float* __restrict__ W1, const float* __restrict__ b1,
          const float* __restrict__ W2, const float* __restrict__ b2,
          float* __restrict__ out)
{
  extern __shared__ _Float16 smem[];
  _Float16* sh   = smem + SH_H;
  _Float16* sWqT = smem + SH_WQT;
  _Float16* sWkT = smem + SH_WKT;
  _Float16* sWvT = smem + SH_WVT;
  _Float16* sW1T = smem + SH_W1T;
  _Float16* sW2T = smem + SH_W2T;
  _Float16* sq   = smem + SH_Q;
  _Float16* sk   = smem + SH_K;
  _Float16* svT  = smem + SH_VT;
  _Float16* so   = smem + SH_O;
  _Float16* so1  = smem + SH_O1;
  _Float16* sp   = smem + SH_P;
  float*    sbias = (float*)(smem + SH_END);   // 5 x 64: bq,bk,bv,b1,b2

  const int bn   = blockIdx.x;            // one block per (b, n)
  const int b    = bn / N_;
  const int n    = bn % N_;
  const int tid  = threadIdx.x;
  const int lane = tid & 31;
  const int wave = tid >> 5;              // 8 waves
  const int r    = lane & 15;
  const int hf   = lane >> 4;

  // ---------- stage inputs into LDS: vectorized float4 -> half4 ----------
  for (int i = tid; i < 64 * 32; i += 256) {          // 64 rows x 32 float4
    int t = i >> 5, j = i & 31;
    size_t base = (((size_t)b * T_ + t) * N_ + n) * D_;
    float4 f4 = (j < 16) ? ((const float4*)(x + base))[j]
                         : ((const float4*)(ste + base))[j - 16];
    v4h h4;
    h4[0] = (_Float16)f4.x; h4[1] = (_Float16)f4.y;
    h4[2] = (_Float16)f4.z; h4[3] = (_Float16)f4.w;
    *(v4h*)(sh + t * 128 + j * 4) = h4;
  }
  // ---------- weights: coalesced global reads, transpose on LDS write ------
  for (int i = tid; i < 64 * 128; i += 256) {         // i = f*64 + dcol
    int f = i >> 6, dcol = i & 63;
    sWqT[dcol * WP + f] = (_Float16)Wq[i];
    sWkT[dcol * WP + f] = (_Float16)Wk[i];
    sWvT[dcol * WP + f] = (_Float16)Wv[i];
  }
  for (int i = tid; i < 64 * 64; i += 256) {
    int f = i >> 6, dcol = i & 63;
    sW1T[dcol * FP + f] = (_Float16)W1[i];
    sW2T[dcol * FP + f] = (_Float16)W2[i];
  }
  if (tid < 64) {
    sbias[tid]       = bq[tid];
    sbias[64 + tid]  = bk[tid];
    sbias[128 + tid] = bv[tid];
    sbias[192 + tid] = b1[tid];
    sbias[256 + tid] = b2[tid];
  }
  __syncthreads();

  // ---------- QKV projections: relu(h @ W + b), 48 tile-jobs over 8 waves ---
  for (int job = wave; job < 48; job += 8) {
    int proj = job / 16, rem = job % 16, mt = rem >> 2, nt = rem & 3;
    const _Float16* WT = (proj == 0) ? sWqT : (proj == 1) ? sWkT : sWvT;
    v8f c = {};
#pragma unroll
    for (int kk = 0; kk < 4; ++kk) {
      v16h a  = load_A(sh + (mt * 16) * 128 + kk * 32, 128);
      v16h bf = load_Bt(WT + (nt * 16) * WP + kk * 32, WP);
      c = wmma16(a, bf, c);
    }
    int ncol = nt * 16 + r;
    float bias = sbias[proj * 64 + ncol];
    if (proj < 2) {
      _Float16* dst = (proj == 0) ? sq : sk;
#pragma unroll
      for (int v = 0; v < 8; ++v) {
        float t = c[v] + bias; t = t > 0.f ? t : 0.f;
        dst[(mt * 16 + v + 8 * hf) * 64 + ncol] = (_Float16)t;
      }
    } else {
      // v stored transposed: vT[feature][time], contiguous in time per lane
      v8h tmp;
#pragma unroll
      for (int v = 0; v < 8; ++v) {
        float t = c[v] + bias; t = t > 0.f ? t : 0.f;
        tmp[v] = (_Float16)t;
      }
      *(v8h*)(svT + ncol * 64 + mt * 16 + 8 * hf) = tmp;
    }
  }
  __syncthreads();

  // ---------- attention: one head per wave, single pass ----------
  const float scale = 0.35355339059327373f;  // 1/sqrt(8)
  {
    const int head = wave;
    _Float16* P = sp + wave * 64 * 64;

    // scores: lower-triangular tiles only (causal); K padded 8 -> 32 with zeros
    for (int mt = 0; mt < 4; ++mt) {
      for (int st = 0; st <= mt; ++st) {
        v16h a = {}, bb = {};
        if (hf == 0) {
          v8h qa = *(const v8h*)(sq + (mt * 16 + r) * 64 + head * 8);
          v8h kb = *(const v8h*)(sk + (st * 16 + r) * 64 + head * 8);
#pragma unroll
          for (int j = 0; j < 8; ++j) { a[j] = qa[j]; bb[j] = kb[j]; }
        }
        v8f c = {};
        c = wmma16(a, bb, c);
#pragma unroll
        for (int v = 0; v < 8; ++v) {
          int trow = mt * 16 + v + 8 * hf;
          int scol = st * 16 + r;
          float s = c[v] * scale;
          if (scol > trow) s = -32767.0f;        // masked; overwritten below
          P[trow * 64 + scol] = (_Float16)s;
        }
      }
    }
    __syncthreads();

    // row softmax over s in [0, row]; zero-fill the causal upper part
    for (int row = lane; row < 64; row += 32) {
      float mx = -1e30f;
      for (int s = 0; s <= row; ++s) {
        float v = (float)P[row * 64 + s];
        mx = v > mx ? v : mx;
      }
      float sum = 0.f;
      for (int s = 0; s <= row; ++s) {
        float e = __expf((float)P[row * 64 + s] - mx);
        sum += e;
        P[row * 64 + s] = (_Float16)e;
      }
      float inv = 1.0f / sum;
      for (int s = 0; s <= row; ++s)
        P[row * 64 + s] = (_Float16)((float)P[row * 64 + s] * inv);
      for (int s = row + 1; s < 64; ++s) P[row * 64 + s] = (_Float16)0.f;
    }
    __syncthreads();

    // out_h = P (64x64) @ v_h (64x8); N padded 8 -> 16, extra cols discarded
    for (int mt = 0; mt < 4; ++mt) {
      v8f c = {};
#pragma unroll
      for (int ks = 0; ks < 2; ++ks) {
        v16h a  = load_A(P + (mt * 16) * 64 + ks * 32, 64);
        v16h bb = {};
        if (r < 8) {
          const _Float16* p = svT + (head * 8 + r) * 64 + ks * 32 + 8 * hf;
          v8h lo = *(const v8h*)p;
          v8h hi = *(const v8h*)(p + 16);
#pragma unroll
          for (int j = 0; j < 8; ++j) { bb[j] = lo[j]; bb[8 + j] = hi[j]; }
        }
        c = wmma16(a, bb, c);
      }
      if (r < 8) {
#pragma unroll
        for (int v = 0; v < 8; ++v)
          so[(mt * 16 + v + 8 * hf) * 64 + head * 8 + r] = (_Float16)c[v];
      }
    }
    __syncthreads();
  }

  // ---------- FFN layer 1: o1 = relu(o @ W1 + b1) ----------
  for (int job = wave; job < 16; job += 8) {
    int mt = job >> 2, nt = job & 3;
    v8f c = {};
#pragma unroll
    for (int kk = 0; kk < 2; ++kk) {
      v16h a  = load_A(so + (mt * 16) * 64 + kk * 32, 64);
      v16h bb = load_Bt(sW1T + (nt * 16) * FP + kk * 32, FP);
      c = wmma16(a, bb, c);
    }
    int ncol = nt * 16 + r;
    float bias = sbias[192 + ncol];
#pragma unroll
    for (int v = 0; v < 8; ++v) {
      float t = c[v] + bias; t = t > 0.f ? t : 0.f;
      so1[(mt * 16 + v + 8 * hf) * 64 + ncol] = (_Float16)t;
    }
  }
  __syncthreads();

  // ---------- FFN layer 2: out = o1 @ W2 + b2 (f32 straight to HBM) ----------
  for (int job = wave; job < 16; job += 8) {
    int mt = job >> 2, nt = job & 3;
    v8f c = {};
#pragma unroll
    for (int kk = 0; kk < 2; ++kk) {
      v16h a  = load_A(so1 + (mt * 16) * 64 + kk * 32, 64);
      v16h bb = load_Bt(sW2T + (nt * 16) * FP + kk * 32, FP);
      c = wmma16(a, bb, c);
    }
    int e = nt * 16 + r;
    float bias = sbias[256 + e];
#pragma unroll
    for (int v = 0; v < 8; ++v) {
      int t = mt * 16 + v + 8 * hf;
      out[(((size_t)b * T_ + t) * N_ + n) * D_ + e] = c[v] + bias;
    }
  }
}

extern "C" void kernel_launch(void* const* d_in, const int* in_sizes, int n_in,
                              void* d_out, int out_size, void* d_ws, size_t ws_size,
                              hipStream_t stream) {
  (void)in_sizes; (void)n_in; (void)out_size; (void)d_ws; (void)ws_size;
  const float* x   = (const float*)d_in[0];
  const float* ste = (const float*)d_in[1];
  const float* Wq  = (const float*)d_in[2];
  const float* bq  = (const float*)d_in[3];
  const float* Wk  = (const float*)d_in[4];
  const float* bk  = (const float*)d_in[5];
  const float* Wv  = (const float*)d_in[6];
  const float* bv  = (const float*)d_in[7];
  const float* W1  = (const float*)d_in[8];
  const float* b1  = (const float*)d_in[9];
  const float* W2  = (const float*)d_in[10];
  const float* b2  = (const float*)d_in[11];
  float* out = (float*)d_out;

  dim3 grid(B_ * N_);   // 4096 blocks: one per (batch, node)
  dim3 block(256);      // 8 waves: one attention head per wave
  ta_kernel<<<grid, block, SMEM_BYTES, stream>>>(x, ste, Wq, bq, Wk, bk, Wv, bv,
                                                 W1, b1, W2, b2, out);
}
